// DeformableTransformerLayer_11665131176620
// MI455X (gfx1250) — matrix operand order repaired and checked by reading, not compiled, not run on hardware
//
#include <hip/hip_runtime.h>
#include <hip/hip_bf16.h>
#include <stdint.h>

// ---------------------------------------------------------------------------
// Problem constants (baked from the reference)
// ---------------------------------------------------------------------------
#define BSZ 4
#define NQ  19320
#define CD  256
#define HH  8
#define LL  3
#define PP  4
#define DD  32
#define NVV 19320
#define MROWS (BSZ * NQ)          // 77280 (== BSZ*NVV as well)

typedef __attribute__((ext_vector_type(16))) _Float16 v16h;
typedef __attribute__((ext_vector_type(8)))  _Float16 v8h;
typedef __attribute__((ext_vector_type(8)))  float    v8f;

// ---------------------------------------------------------------------------
// CDNA5 async global->LDS copy (ASYNCcnt) helpers.
// LDS operand: low 32 bits of the generic shared pointer == LDS byte address.
// ---------------------------------------------------------------------------
static __device__ __forceinline__
void async_cp16(void* lds, const void* gptr) {
    asm volatile("global_load_async_to_lds_b128 %0, %1, off"
                 :: "v"((uint32_t)(uintptr_t)lds),
                    "v"((uint64_t)(uintptr_t)gptr)
                 : "memory");
}
#define WAIT_ASYNC(n) asm volatile("s_wait_asynccnt " #n ::: "memory")

// ---------------------------------------------------------------------------
// Elementwise f32 -> f16
// ---------------------------------------------------------------------------
__global__ __launch_bounds__(256)
void f32tof16_kernel(const float* __restrict__ x, _Float16* __restrict__ y, int n)
{
    const int i = blockIdx.x * 256 + threadIdx.x;
    if (i < n) y[i] = (_Float16)x[i];
}

// ---------------------------------------------------------------------------
// Weight convert + transpose: W (K,N) f32  ->  Wt (N,K) f16
// ---------------------------------------------------------------------------
__global__ __launch_bounds__(256)
void wcvt_kernel(const float* __restrict__ W, _Float16* __restrict__ Wt, int K, int N)
{
    const int i = blockIdx.x * 256 + threadIdx.x;
    if (i >= K * N) return;
    const int k = i / N, n = i - k * N;           // coalesced read over n
    Wt[(size_t)n * K + k] = (_Float16)W[i];
}

// ---------------------------------------------------------------------------
// LayerNorm. One block per row (256 threads = C). Wave shfl_xor reductions.
// y_ln (f32, optional) = LN(x); y_q (f16, optional) = LN(x) [+ pos].
// ---------------------------------------------------------------------------
__global__ __launch_bounds__(256)
void ln_kernel(const float* __restrict__ x, const float* __restrict__ w,
               const float* __restrict__ b, const float* __restrict__ pos,
               float* y_ln, _Float16* y_q)
{
    const int row  = blockIdx.x;
    const int t    = threadIdx.x;
    const int lane = t & 31;
    const int wv   = t >> 5;
    const size_t base = (size_t)row * CD;
    const float v = x[base + t];

    __shared__ float part[8];

    float s = v;
    #pragma unroll
    for (int o = 16; o > 0; o >>= 1) s += __shfl_xor(s, o, 32);
    if (lane == 0) part[wv] = s;
    __syncthreads();
    float tot = 0.0f;
    #pragma unroll
    for (int i = 0; i < 8; ++i) tot += part[i];
    const float mean = tot * (1.0f / CD);
    __syncthreads();

    const float d = v - mean;
    s = d * d;
    #pragma unroll
    for (int o = 16; o > 0; o >>= 1) s += __shfl_xor(s, o, 32);
    if (lane == 0) part[wv] = s;
    __syncthreads();
    tot = 0.0f;
    #pragma unroll
    for (int i = 0; i < 8; ++i) tot += part[i];
    const float var = tot * (1.0f / CD);

    const float ln = d * rsqrtf(var + 1e-5f) * w[t] + b[t];
    if (y_ln) y_ln[base + t] = ln;
    if (y_q)  y_q[base + t]  = (_Float16)(pos ? ln + pos[base + t] : ln);
}

// ---------------------------------------------------------------------------
// WMMA GEMM: Y[M,N] = act(X[M,K] @ W + bias) + res0 + res1
//   X  : f16 (M,K) row-major (produced by earlier stages)
//   Wt : f16 (N,K) row-major (pre-transposed weights)
// Tiles (A: BMxBK, B: BNxBK, both K-contiguous) are staged into LDS with
// double-buffered GLOBAL_LOAD_ASYNC_TO_LDS_B128 (3 async instr / wave /
// stage, uniform EXEC -> fixed s_wait_asynccnt). 8 waves; wave (wm,wn) owns
// two 16x16 tiles at rows {16*wm, 64+16*wm}, col 16*wn, sharing one B
// fragment: 2x v_wmma_f32_16x16x32_f16 per K-step.
// ACT: 0 = identity, 1 = exact GELU.  F16OUT: store f16 instead of f32.
// ---------------------------------------------------------------------------
#define BM 128
#define BN 32
#define BK 32

static __device__ __forceinline__ v16h cat8(v8h lo, v8h hi) {
    return __builtin_shufflevector(lo, hi, 0, 1, 2, 3, 4, 5, 6, 7,
                                           8, 9, 10, 11, 12, 13, 14, 15);
}

template<int ACT, int F16OUT>
__global__ __launch_bounds__(256)
void wmma_gemm_kernel(const _Float16* __restrict__ X, const _Float16* __restrict__ Wt,
                      const float* __restrict__ bias,
                      const float* res0, const float* res1, void* Yv,
                      int Mtot, int K, int N)
{
    __shared__ __attribute__((aligned(16))) _Float16 As[2][BM][BK];  // 16 KB
    __shared__ __attribute__((aligned(16))) _Float16 Bt[2][BN][BK];  //  4 KB

    const int m0   = blockIdx.x * BM;
    const int n0   = blockIdx.y * BN;
    const int t    = threadIdx.x;
    const int lane = t & 31;
    const int w    = t >> 5;
    const int wm   = w & 3;
    const int wn   = w >> 2;
    const int lr   = lane & 15;
    const int hi   = lane >> 4;
    const int khA  = hi << 3;              // A K-interleave (8-granular)
    const int khB  = hi << 3;              // B K-interleave (8-granular)

    // A: 128 rows x 64B = 512 x 16B chunks; thread t copies chunks t, t+256.
    const int a0r = t >> 2;                // 0..63
    const int a1r = a0r + 64;              // 64..127
    const int ac  = (t & 3) << 3;          // halves offset 0/8/16/24
    // B: 32 rows x 64B = 128 chunks; threads t and t+128 duplicate chunk t&127.
    const int brr = (t & 127) >> 2;        // n row 0..31
    const int bcc = (t & 3) << 3;

    // clamp edge rows (garbage only reaches guarded, never-stored outputs)
    int gm0 = m0 + a0r; if (gm0 >= Mtot) gm0 = Mtot - 1;
    int gm1 = m0 + a1r; if (gm1 >= Mtot) gm1 = Mtot - 1;
    const _Float16* srcA0 = X + (size_t)gm0 * K + ac;
    const _Float16* srcA1 = X + (size_t)gm1 * K + ac;
    const _Float16* srcB  = Wt + (size_t)(n0 + brr) * K + bcc;

    const int nk = K / BK;
    v8f acc0 = {}, acc1 = {};

    // prologue: stage 0 into buffer 0
    async_cp16(&As[0][a0r][ac], srcA0);
    async_cp16(&As[0][a1r][ac], srcA1);
    async_cp16(&Bt[0][brr][bcc], srcB);

    for (int kt = 0; kt < nk; ++kt) {
        const int buf = kt & 1;
        if (kt + 1 < nk) {
            const int k1 = (kt + 1) * BK;
            async_cp16(&As[buf ^ 1][a0r][ac], srcA0 + k1);
            async_cp16(&As[buf ^ 1][a1r][ac], srcA1 + k1);
            async_cp16(&Bt[buf ^ 1][brr][bcc], srcB + k1);
            WAIT_ASYNC(3);                 // current stage done, next in flight
        } else {
            WAIT_ASYNC(0);
        }
        __syncthreads();

        const int col = wn * 16 + lr;
        const v16h b = cat8(*(const v8h*)&Bt[buf][col][khB],
                            *(const v8h*)&Bt[buf][col][khB + 16]);

        const int rowA0 = wm * 16 + lr;
        const v16h a0 = cat8(*(const v8h*)&As[buf][rowA0][khA],
                             *(const v8h*)&As[buf][rowA0][khA + 16]);
        acc0 = __builtin_amdgcn_wmma_f32_16x16x32_f16(
            false, a0, false, b, (short)0, acc0, false, false);

        const int rowA1 = 64 + wm * 16 + lr;
        const v16h a1 = cat8(*(const v8h*)&As[buf][rowA1][khA],
                             *(const v8h*)&As[buf][rowA1][khA + 16]);
        acc1 = __builtin_amdgcn_wmma_f32_16x16x32_f16(
            false, a1, false, b, (short)0, acc1, false, false);

        __syncthreads();                   // safe to overwrite buf next iter
    }

    // ---- epilogue: bias (+GELU) (+residuals) ----
    const int n  = n0 + wn * 16 + lr;
    const int mb = m0 + wm * 16 + (hi << 3);
    const float bn = bias ? bias[n] : 0.0f;
    #pragma unroll
    for (int r = 0; r < 8; ++r) {
        #pragma unroll
        for (int half = 0; half < 2; ++half) {
            const int m = mb + 64 * half + r;
            if (m < Mtot) {
                float v = (half ? acc1[r] : acc0[r]) + bn;
                if (ACT == 1) v = 0.5f * v * (1.0f + erff(v * 0.70710678118654752f));
                const size_t idx = (size_t)m * N + n;
                if (res0) v += res0[idx];
                if (res1) v += res1[idx];
                if (F16OUT) ((_Float16*)Yv)[idx] = (_Float16)v;
                else        ((float*)Yv)[idx]    = v;
            }
        }
    }
}

// ---------------------------------------------------------------------------
// Softmax over the 12 (L*P) logits per (row, h), in place on (M, 96) f32.
// ---------------------------------------------------------------------------
__global__ __launch_bounds__(256)
void softmax12_kernel(float* a, int total)
{
    const int g = blockIdx.x * 256 + threadIdx.x;
    if (g >= total) return;
    const int row = g >> 3;
    const int h   = g & 7;
    float* p = a + (size_t)row * (HH * LL * PP) + h * (LL * PP);
    float mx = p[0];
    #pragma unroll
    for (int i = 1; i < 12; ++i) mx = fmaxf(mx, p[i]);
    float e[12], s = 0.0f;
    #pragma unroll
    for (int i = 0; i < 12; ++i) { e[i] = __expf(p[i] - mx); s += e[i]; }
    const float inv = 1.0f / s;
    #pragma unroll
    for (int i = 0; i < 12; ++i) p[i] = e[i] * inv;
}

// ---------------------------------------------------------------------------
// MSDA core. One wave32 per (b, q, h); lane = d. Values are f16 (64 B
// contiguous per tap per wave); math in f32; output f16 for the next GEMM.
// ---------------------------------------------------------------------------
__global__ __launch_bounds__(256)
void msda_kernel(const _Float16* __restrict__ val, const float* __restrict__ off,
                 const float* __restrict__ aw, const float* __restrict__ ref,
                 _Float16* __restrict__ out)
{
    const int g = blockIdx.x * 8 + (threadIdx.x >> 5);
    if (g >= MROWS * HH) return;
    const int lane = threadIdx.x & 31;           // = d
    const int row  = g >> 3;                     // b*NQ + q
    const int h    = g & 7;
    const int b    = row / NQ;

    const int Hs[LL] = {92, 46, 23};
    const int Ws[LL] = {160, 80, 40};
    const int St[LL] = {0, 14720, 18400};

    const _Float16* vbase = val + (size_t)b * NVV * CD + h * DD + lane;
    float acc = 0.0f;

    #pragma unroll
    for (int l = 0; l < LL; ++l) {
        const float Wl = (float)Ws[l], Hl = (float)Hs[l];
        const float rx = ref[((size_t)row * LL + l) * 2 + 0];
        const float ry = ref[((size_t)row * LL + l) * 2 + 1];
        #pragma unroll
        for (int p = 0; p < PP; ++p) {
            const size_t oi = (size_t)row * (HH * LL * PP * 2)
                            + (((h * LL + l) * PP + p) * 2);
            const float x = rx * Wl + off[oi + 0] - 0.5f;
            const float y = ry * Hl + off[oi + 1] - 0.5f;
            const float x0f = floorf(x), y0f = floorf(y);
            const float wx = x - x0f, wy = y - y0f;
            const int x0 = (int)x0f, y0 = (int)y0f;
            const float a = aw[(size_t)row * (HH * LL * PP) + h * (LL * PP) + l * PP + p];

            const float cw[4] = { (1.0f - wx) * (1.0f - wy), wx * (1.0f - wy),
                                  (1.0f - wx) * wy,          wx * wy };
            const int cx[4] = { x0, x0 + 1, x0,     x0 + 1 };
            const int cy[4] = { y0, y0,     y0 + 1, y0 + 1 };
            #pragma unroll
            for (int c = 0; c < 4; ++c) {
                const int xi = cx[c], yi = cy[c];
                if (xi >= 0 && xi < Ws[l] && yi >= 0 && yi < Hs[l]) {
                    const int pix = St[l] + yi * Ws[l] + xi;
                    acc += a * cw[c] * (float)vbase[(size_t)pix * CD];
                }
            }
        }
    }
    out[(size_t)row * CD + h * DD + lane] = (_Float16)acc;
}

// ---------------------------------------------------------------------------
// Launch
// ---------------------------------------------------------------------------
extern "C" void kernel_launch(void* const* d_in, const int* in_sizes, int n_in,
                              void* d_out, int out_size, void* d_ws, size_t ws_size,
                              hipStream_t stream)
{
    (void)in_sizes; (void)n_in; (void)out_size; (void)ws_size;

    const float* query     = (const float*)d_in[0];
    const float* value     = (const float*)d_in[1];
    const float* query_pos = (const float*)d_in[2];
    const float* ref_pts   = (const float*)d_in[3];
    const float* ln1_w = (const float*)d_in[6];
    const float* ln1_b = (const float*)d_in[7];
    const float* ln2_w = (const float*)d_in[8];
    const float* ln2_b = (const float*)d_in[9];
    const float* W_off  = (const float*)d_in[10];
    const float* b_off  = (const float*)d_in[11];
    const float* W_attn = (const float*)d_in[12];
    const float* b_attn = (const float*)d_in[13];
    const float* W_val  = (const float*)d_in[14];
    const float* b_val  = (const float*)d_in[15];
    const float* W_out  = (const float*)d_in[16];
    const float* b_out  = (const float*)d_in[17];
    const float* W_ffn1 = (const float*)d_in[18];
    const float* b_ffn1 = (const float*)d_in[19];
    const float* W_ffn2 = (const float*)d_in[20];
    const float* b_ffn2 = (const float*)d_in[21];
    float* out = (float*)d_out;

    // ---- workspace layout ----
    char* wsb = (char*)d_ws;
    size_t o = 0;
    auto allocf = [&](size_t n) { float* p = (float*)(wsb + o); o += n * 4; return p; };
    auto alloch = [&](size_t n) { _Float16* p = (_Float16*)(wsb + o); o += n * 2; return p; };

    float*    q1     = allocf((size_t)MROWS * CD);     // LN1(query), residual
    float*    offb   = allocf((size_t)MROWS * 192);    // sampling offsets
    float*    attn   = allocf((size_t)MROWS * 96);     // logits -> softmax
    _Float16* qh     = alloch((size_t)MROWS * CD);     // (q1+pos) f16
    _Float16* valueh = alloch((size_t)MROWS * CD);     // raw value f16
    _Float16* valh   = alloch((size_t)MROWS * CD);     // value projection f16
    _Float16* msdah  = alloch((size_t)MROWS * CD);     // MSDA out f16
    _Float16* q2h    = alloch((size_t)MROWS * CD);     // LN2(x) f16
    _Float16* ffnh   = alloch((size_t)MROWS * 1024);   // FFN hidden f16
    _Float16* WtOff  = alloch((size_t)CD * 192);
    _Float16* WtAttn = alloch((size_t)CD * 96);
    _Float16* WtVal  = alloch((size_t)CD * CD);
    _Float16* WtOut  = alloch((size_t)CD * CD);
    _Float16* WtF1   = alloch((size_t)CD * 1024);
    _Float16* WtF2   = alloch((size_t)1024 * CD);

    const dim3 blk(256);
    const int GM = (MROWS + BM - 1) / BM;              // 604
    auto cdiv = [](int a, int b) { return (a + b - 1) / b; };

    // 0) weight convert+transpose (tiny) and value f32->f16
    wcvt_kernel<<<cdiv(CD * 192, 256), blk, 0, stream>>>(W_off, WtOff, CD, 192);
    wcvt_kernel<<<cdiv(CD * 96, 256), blk, 0, stream>>>(W_attn, WtAttn, CD, 96);
    wcvt_kernel<<<cdiv(CD * CD, 256), blk, 0, stream>>>(W_val, WtVal, CD, CD);
    wcvt_kernel<<<cdiv(CD * CD, 256), blk, 0, stream>>>(W_out, WtOut, CD, CD);
    wcvt_kernel<<<cdiv(CD * 1024, 256), blk, 0, stream>>>(W_ffn1, WtF1, CD, 1024);
    wcvt_kernel<<<cdiv(1024 * CD, 256), blk, 0, stream>>>(W_ffn2, WtF2, 1024, CD);
    f32tof16_kernel<<<cdiv(MROWS * CD, 256), blk, 0, stream>>>(value, valueh, MROWS * CD);

    // 1) q1 = LN1(query) [f32]; qh = q1 + query_pos [f16]
    ln_kernel<<<MROWS, blk, 0, stream>>>(query, ln1_w, ln1_b, query_pos, q1, qh);
    // 2) offb = qh @ W_off + b_off                   (M x 192, f32)
    wmma_gemm_kernel<0, 0><<<dim3(GM, 192 / BN), blk, 0, stream>>>(
        qh, WtOff, b_off, nullptr, nullptr, offb, MROWS, CD, 192);
    // 3) attn = qh @ W_attn + b_attn                 (M x 96, f32)
    wmma_gemm_kernel<0, 0><<<dim3(GM, 96 / BN), blk, 0, stream>>>(
        qh, WtAttn, b_attn, nullptr, nullptr, attn, MROWS, CD, 96);
    // 4) softmax over L*P per (row, h), in place
    softmax12_kernel<<<cdiv(MROWS * HH, 256), blk, 0, stream>>>(attn, MROWS * HH);
    // 5) valh = value @ W_val + b_val                (M x 256, f16)
    wmma_gemm_kernel<0, 1><<<dim3(GM, CD / BN), blk, 0, stream>>>(
        valueh, WtVal, b_val, nullptr, nullptr, valh, MROWS, CD, CD);
    // 6) MSDA bilinear gather-accumulate (f16 values -> f16 out)
    msda_kernel<<<cdiv(MROWS * HH, 8), blk, 0, stream>>>(valh, offb, attn, ref_pts, msdah);
    // 7) x = msdah @ W_out + b_out + query + q1  -> d_out (f32)
    wmma_gemm_kernel<0, 0><<<dim3(GM, CD / BN), blk, 0, stream>>>(
        msdah, WtOut, b_out, query, q1, out, MROWS, CD, CD);
    // 8) q2h = LN2(x) [f16]
    ln_kernel<<<MROWS, blk, 0, stream>>>(out, ln2_w, ln2_b, nullptr, nullptr, q2h);
    // 9) ffnh = GELU(q2h @ W_ffn1 + b_ffn1)          (M x 1024, f16)
    wmma_gemm_kernel<1, 1><<<dim3(GM, 1024 / BN), blk, 0, stream>>>(
        q2h, WtF1, b_ffn1, nullptr, nullptr, ffnh, MROWS, CD, 1024);
    // 10) d_out = x + ffnh @ W_ffn2 + b_ffn2 (in-place residual, f32)
    wmma_gemm_kernel<0, 0><<<dim3(GM, CD / BN), blk, 0, stream>>>(
        ffnh, WtF2, b_ffn2, out, nullptr, out, MROWS, 1024, CD);
}
